// Block_4956392259615
// MI455X (gfx1250) — matrix operand order repaired and checked by reading, not compiled, not run on hardware
//
#include <hip/hip_runtime.h>
#include <hip/hip_fp16.h>

typedef __attribute__((ext_vector_type(16))) _Float16 v16h;
typedef __attribute__((ext_vector_type(8)))  float    v8f;

#define N_NODES 50000
#define N_EDGES 312500
#define WIDTH   256
#define LN_EPS  1e-5f

// ---------------- degree kernels ----------------
__global__ void deg_init(int* __restrict__ deg) {
    int i = blockIdx.x * blockDim.x + threadIdx.x;
    if (i < N_NODES) deg[i] = 1;            // self-loop
}

__global__ void deg_accum(const int* __restrict__ ei, int* __restrict__ deg) {
    int e = blockIdx.x * blockDim.x + threadIdx.x;
    if (e < N_EDGES) {
        int dst = ei[N_EDGES + e];          // edge_index[1][e]
        atomicAdd(&deg[dst], 1);
    }
}

__global__ void deg_rsqrt(const int* __restrict__ deg, float* __restrict__ dinv) {
    int i = blockIdx.x * blockDim.x + threadIdx.x;
    if (i < N_NODES) dinv[i] = rsqrtf((float)deg[i]);
}

// ---------------- WMMA GEMM: h = x @ W^T (f16 out), seed out with self-loop ----------------
// Block: 256 threads = 8 waves. Block covers 16 rows x 256 cols. Wave w covers cols [32w, 32w+32).
__global__ void gemm_wmma(const float* __restrict__ x, const float* __restrict__ W,
                          const float* __restrict__ dinv,
                          _Float16* __restrict__ h, float* __restrict__ out) {
    const int row0 = blockIdx.x * 16;
    const int wave = threadIdx.x >> 5;
    const int lane = threadIdx.x & 31;
    const int nidx = lane & 15;       // M for A-tile, N for B-tile
    const int half = lane >> 4;       // K-half selector
    const int col0 = wave * 32;

    const float* __restrict__ xrow = x + (size_t)(row0 + nidx) * WIDTH;
    const float* __restrict__ w0   = W + (size_t)(col0 + nidx) * WIDTH;
    const float* __restrict__ w1   = W + (size_t)(col0 + 16 + nidx) * WIDTH;

    v8f acc0 = {};
    v8f acc1 = {};

    #pragma unroll
    for (int k0 = 0; k0 < WIDTH; k0 += 32) {
        v16h a, b0, b1;
        // A 16x32 f16, contiguous k order chosen here: lanes 0-15 => K 0..15; lanes 16-31 => K 16..31
        #pragma unroll
        for (int e = 0; e < 16; ++e) a[e] = (_Float16)xrow[k0 + half * 16 + e];
        // B 32x16 f16, same contiguous k order as A: lanes 0-15 hold K=0..15, lanes 16-31 hold K=16..31 (col = lane&15)
        #pragma unroll
        for (int e = 0; e < 16; ++e) b0[e] = (_Float16)w0[k0 + half * 16 + e];
        #pragma unroll
        for (int e = 0; e < 16; ++e) b1[e] = (_Float16)w1[k0 + half * 16 + e];

        acc0 = __builtin_amdgcn_wmma_f32_16x16x32_f16(false, a, false, b0, (short)0, acc0, false, false);
        acc1 = __builtin_amdgcn_wmma_f32_16x16x32_f16(false, a, false, b1, (short)0, acc1, false, false);
    }

    // C/D layout: VGPR r, lanes 0-15 -> M=r, lanes 16-31 -> M=8+r; N = lane&15
    #pragma unroll
    for (int r = 0; r < 8; ++r) {
        const int m  = row0 + half * 8 + r;
        const float dv = dinv[m];
        const float s  = dv * dv;                     // self-loop norm
        const int c0 = col0 + nidx;
        const int c1 = col0 + 16 + nidx;
        h[(size_t)m * WIDTH + c0]   = (_Float16)acc0[r];
        h[(size_t)m * WIDTH + c1]   = (_Float16)acc1[r];
        out[(size_t)m * WIDTH + c0] = s * acc0[r];    // fully initializes d_out
        out[(size_t)m * WIDTH + c1] = s * acc1[r];
    }
}

// ---------------- edge scatter: out[dst] += dinv[src]*dinv[dst]*h[src] ----------------
// One wave32 per edge, 8 contiguous f16 per lane (b128 gather), f32 atomics into L2.
__global__ void edge_scatter(const int* __restrict__ ei, const float* __restrict__ dinv,
                             const _Float16* __restrict__ h, float* __restrict__ out) {
    const int wave = threadIdx.x >> 5;
    const int lane = threadIdx.x & 31;
    const int e = blockIdx.x * 8 + wave;
    if (e >= N_EDGES) return;
    const int src = ei[e];
    const int dst = ei[N_EDGES + e];
    const float norm = dinv[src] * dinv[dst];
    const _Float16* __restrict__ hs = h   + (size_t)src * WIDTH;
    float* __restrict__          od = out + (size_t)dst * WIDTH;

    _Float16 v[8];
    #pragma unroll
    for (int j = 0; j < 8; ++j) v[j] = hs[lane * 8 + j];
    #pragma unroll
    for (int j = 0; j < 8; ++j)
        unsafeAtomicAdd(&od[lane * 8 + j], norm * (float)v[j]);
}

// ---------------- bias + LayerNorm + ReLU, in place on d_out ----------------
// One wave32 per row, 8 features per lane, shuffle reduction.
__global__ void ln_relu(float* __restrict__ out, const float* __restrict__ bias,
                        const float* __restrict__ gamma, const float* __restrict__ beta) {
    const int wave = threadIdx.x >> 5;
    const int lane = threadIdx.x & 31;
    const int row = blockIdx.x * 8 + wave;
    if (row >= N_NODES) return;
    float* __restrict__ o = out + (size_t)row * WIDTH;

    float v[8];
    float s = 0.f, s2 = 0.f;
    #pragma unroll
    for (int j = 0; j < 8; ++j) {
        const int c = lane * 8 + j;
        v[j] = o[c] + bias[c];
        s  += v[j];
        s2 += v[j] * v[j];
    }
    #pragma unroll
    for (int off = 16; off > 0; off >>= 1) {
        s  += __shfl_xor(s,  off, 32);
        s2 += __shfl_xor(s2, off, 32);
    }
    const float mean = s * (1.0f / WIDTH);
    const float var  = s2 * (1.0f / WIDTH) - mean * mean;
    const float rstd = rsqrtf(var + LN_EPS);
    #pragma unroll
    for (int j = 0; j < 8; ++j) {
        const int c = lane * 8 + j;
        float y = (v[j] - mean) * rstd * gamma[c] + beta[c];
        o[c] = y > 0.f ? y : 0.f;
    }
}

// ---------------- launcher ----------------
extern "C" void kernel_launch(void* const* d_in, const int* in_sizes, int n_in,
                              void* d_out, int out_size, void* d_ws, size_t ws_size,
                              hipStream_t stream) {
    (void)in_sizes; (void)n_in; (void)out_size; (void)ws_size;
    const float* x     = (const float*)d_in[0];
    const int*   ei    = (const int*)  d_in[1];
    const float* W     = (const float*)d_in[2];
    const float* bias  = (const float*)d_in[3];
    const float* gamma = (const float*)d_in[4];
    const float* beta  = (const float*)d_in[5];
    float* out = (float*)d_out;

    char* ws = (char*)d_ws;
    int*      degi = (int*)ws;                       // 200 KB
    float*    dinv = (float*)(ws + 256 * 1024);      // 200 KB
    _Float16* h    = (_Float16*)(ws + 512 * 1024);   // 25.6 MB

    deg_init  <<<(N_NODES + 255) / 256, 256, 0, stream>>>(degi);
    deg_accum <<<(N_EDGES + 255) / 256, 256, 0, stream>>>(ei, degi);
    deg_rsqrt <<<(N_NODES + 255) / 256, 256, 0, stream>>>(degi, dinv);
    gemm_wmma <<<N_NODES / 16, 256, 0, stream>>>(x, W, dinv, h, out);
    edge_scatter<<<(N_EDGES + 7) / 8, 256, 0, stream>>>(ei, dinv, h, out);
    ln_relu   <<<(N_NODES + 7) / 8, 256, 0, stream>>>(out, bias, gamma, beta);
}
